// DecisionTransformerGPT2Attention_59133109732142
// MI455X (gfx1250) — matrix operand order repaired and checked by reading, not compiled, not run on hardware
//
#include <hip/hip_runtime.h>

typedef _Float16 h8_t  __attribute__((ext_vector_type(8)));
typedef _Float16 h16_t __attribute__((ext_vector_type(16)));
typedef float    f8_t  __attribute__((ext_vector_type(8)));
typedef int      v4i_t __attribute__((ext_vector_type(4)));

#define B_   2
#define S_   2048
#define D_   1024
#define H_   16
#define HD_  64
#define WIN_ 512
#define M_   (B_ * S_)      // 4096

#if defined(__has_builtin)
#if __has_builtin(__builtin_amdgcn_global_load_async_to_lds_b128) && \
    __has_builtin(__builtin_amdgcn_s_wait_asynccnt)
#define HAVE_ASYNC_LDS 1
#endif
#endif

union F16Frag { h16_t v; h8_t part[2]; _Float16 e[16]; };

// A-fragment (16-bit A 16x32 layout): per lane, values at p[0..15]
static __device__ inline h16_t load_a_frag(const _Float16* p) {
  F16Frag a;
  a.part[0] = *(const h8_t*)(p);
  a.part[1] = *(const h8_t*)(p + 8);
  return a.v;
}

static __device__ inline f8_t wmma_f16(h16_t a, h16_t b, f8_t c) {
  return __builtin_amdgcn_wmma_f32_16x16x32_f16(false, a, false, b, (short)0, c,
                                                false, false);
}

// Copy 16 bytes global -> LDS (async path if available)
static __device__ inline void stage16(const _Float16* gsrc, _Float16* ldst) {
#ifdef HAVE_ASYNC_LDS
  __builtin_amdgcn_global_load_async_to_lds_b128(
      (__attribute__((address_space(1))) v4i_t*)gsrc,
      (__attribute__((address_space(3))) v4i_t*)ldst, 0, 0);
#else
  *(h8_t*)ldst = *(const h8_t*)gsrc;
#endif
}

static __device__ inline void stage_wait() {
#ifdef HAVE_ASYNC_LDS
  __builtin_amdgcn_s_wait_asynccnt(0);
#endif
}

// ---------------- conversion / layout kernels ----------------

__global__ void k_cvt(const float* __restrict__ src, _Float16* __restrict__ dst, int n) {
  int i = blockIdx.x * blockDim.x + threadIdx.x;
  if (i < n) dst[i] = (_Float16)src[i];
}

// WT[n*K + k] = W[k*N + n]   (f32 -> f16)
__global__ void k_transpose_cvt(const float* __restrict__ W, _Float16* __restrict__ WT,
                                int K, int N) {
  int i = blockIdx.x * blockDim.x + threadIdx.x;
  if (i >= K * N) return;
  int k = i % K, n = i / K;
  WT[(size_t)n * K + k] = (_Float16)W[(size_t)k * N + n];
}

// In-place RoPE on Q (part 0) and K (part 1) of QKV[(part,b,h,s,hd)]
__global__ void k_rope(_Float16* __restrict__ QKV) {
  unsigned i = blockIdx.x * blockDim.x + threadIdx.x;   // 2*2*16*2048*32 threads
  int j = i & 31;
  unsigned row = i >> 5;                 // ((part*B+b)*H+h)*S + s
  int s = row & (S_ - 1);
  size_t base = (size_t)row * HD_;
  float inv_freq = __expf(-((float)(2 * j) / (float)HD_) * 8.29404964f); // ln(4000)
  float ang = (float)s * inv_freq;
  float cs = __cosf(ang), sn = __sinf(ang);
  float x1 = (float)QKV[base + j];
  float x2 = (float)QKV[base + j + 32];
  QKV[base + j]      = (_Float16)(x1 * cs - x2 * sn);
  QKV[base + j + 32] = (_Float16)(x2 * cs + x1 * sn);
}

// VT[((b*H+h)*HD + d)*S + s] = V[(b,h,s,d)]  (V = part 2 of QKV)
__global__ void k_vt(const _Float16* __restrict__ QKV, _Float16* __restrict__ VT) {
  unsigned i = blockIdx.x * blockDim.x + threadIdx.x;   // 2*16*2048*64 threads
  int d = i & (HD_ - 1);
  unsigned t = i >> 6;
  int s = t & (S_ - 1);
  unsigned bh = t >> 11;                 // 0..31
  _Float16 v = QKV[(((size_t)(2 * B_ * H_) + bh) * S_ + s) * HD_ + d];
  VT[((size_t)bh * HD_ + d) * S_ + s] = v;
}

// ---------------- GEMM 1: QKV = X @ Wa + ba ----------------
// Block = 256 thr (8 waves) -> 128x64 tile. B tile (64n x 32k) double-buffered
// in LDS via async-to-LDS: copy for chunk kb+32 is in flight while WMMAs
// consume chunk kb; s_wait_asynccnt + barrier only at the end of the iteration.

__global__ void __launch_bounds__(256) k_gemm_qkv(const _Float16* __restrict__ X,
                                                  const _Float16* __restrict__ WaT,
                                                  const float* __restrict__ bias,
                                                  _Float16* __restrict__ QKV) {
  __shared__ _Float16 Bsh[2][64 * 32];                // ping-pong [n][k]
  const int NBT = (3 * D_) / 64;                      // 48
  int bm = blockIdx.x / NBT, bn = blockIdx.x % NBT;
  int n0 = bn * 64;
  int t = threadIdx.x;
  int lane = t & 31;
  int hlf = lane >> 4, l16 = lane & 15;
  int m0 = bm * 128 + (t >> 5) * 16;

  const _Float16* bsrc = WaT + (size_t)(n0 + (t >> 2)) * D_ + (t & 3) * 8;
  _Float16* bdst0 = &Bsh[0][t * 8];
  _Float16* bdst1 = &Bsh[1][t * 8];
  const _Float16* arow = X + (size_t)(m0 + l16) * D_ + hlf * 16;

  stage16(bsrc, bdst0);                               // prologue: chunk 0
  stage_wait();
  __syncthreads();

  f8_t acc[4] = {};
  for (int kb = 0; kb < D_; kb += 32) {
    int cur = (kb >> 5) & 1;
    if (kb + 32 < D_) stage16(bsrc + kb + 32, cur ? bdst0 : bdst1);  // async fill next
    const _Float16* Bcur = Bsh[cur];
    h16_t a = load_a_frag(arow + kb);
#pragma unroll
    for (int j = 0; j < 4; ++j) {
      h16_t b = *(const h16_t*)&Bcur[(j * 16 + l16) * 32 + hlf * 16];
      acc[j] = wmma_f16(a, b, acc[j]);
    }
    stage_wait();                                     // next chunk landed
    __syncthreads();
  }

#pragma unroll
  for (int j = 0; j < 4; ++j) {
    int n = n0 + j * 16 + l16;
    float bn2 = bias[n];
    int part = n >> 10, rem = n & (D_ - 1), hh = rem >> 6, dd = rem & (HD_ - 1);
#pragma unroll
    for (int r = 0; r < 8; ++r) {
      int m = m0 + r + hlf * 8;
      int bb = m >> 11, s = m & (S_ - 1);
      QKV[((((size_t)part * B_ + bb) * H_ + hh) * S_ + s) * HD_ + dd] =
          (_Float16)(acc[j][r] + bn2);
    }
  }
}

// ---------------- flash attention (window 512), one wave per 16-query tile ----------------

__global__ void __launch_bounds__(128) k_attn(const _Float16* __restrict__ QKV,
                                              const _Float16* __restrict__ VT,
                                              _Float16* __restrict__ CTX) {
  __shared__ _Float16 Psh[4][16 * 32];
  int wib  = threadIdx.x >> 5;
  int gw   = blockIdx.x * 4 + wib;        // 4096 waves = B*H*(S/16)
  int lane = threadIdx.x & 31;
  int hlf = lane >> 4, l16 = lane & 15;
  int qt = gw & 127;
  int bh = gw >> 7;                        // 0..31
  int b = bh >> 4, h = bh & 15;
  int q0 = qt * 16;

  const _Float16* Qbase = QKV + (((size_t)(0 * B_ + b) * H_ + h) * S_) * HD_;
  const _Float16* Kbase = QKV + (((size_t)(1 * B_ + b) * H_ + h) * S_) * HD_;
  const _Float16* Vbase = VT  + ((size_t)bh * HD_) * S_;

  const _Float16* qp = Qbase + (size_t)(q0 + l16) * HD_;
  h16_t aQ0 = load_a_frag(qp + hlf * 16);
  h16_t aQ1 = load_a_frag(qp + 32 + hlf * 16);

  float mx[8], lsum[8];
  f8_t o0 = {}, o1 = {}, o2 = {}, o3 = {};
#pragma unroll
  for (int r = 0; r < 8; ++r) { mx[r] = -3.0e38f; lsum[r] = 0.0f; }

  int kfirst = q0 - (WIN_ - 1);
  if (kfirst < 0) kfirst = 0;
  kfirst &= ~31;
  int klast = q0 + 15;

  for (int k0 = kfirst; k0 <= klast; k0 += 32) {
    const _Float16* kp0 = Kbase + (size_t)(k0 + l16) * HD_;
    const _Float16* kp1 = Kbase + (size_t)(k0 + 16 + l16) * HD_;
    h16_t b00 = *(const h16_t*)(kp0 + hlf * 16);
    h16_t b01 = *(const h16_t*)(kp0 + 32 + hlf * 16);
    h16_t b10 = *(const h16_t*)(kp1 + hlf * 16);
    h16_t b11 = *(const h16_t*)(kp1 + 32 + hlf * 16);
    f8_t c0 = {}, c1 = {};
    c0 = wmma_f16(aQ0, b00, c0);
    c0 = wmma_f16(aQ1, b01, c0);
    c1 = wmma_f16(aQ0, b10, c1);
    c1 = wmma_f16(aQ1, b11, c1);

    const float scale = 0.125f;   // 1/sqrt(64)
#pragma unroll
    for (int r = 0; r < 8; ++r) {
      int qi = q0 + r + hlf * 8;
      int ka = k0 + l16, kb2 = k0 + 16 + l16;
      bool a0 = (ka  <= qi) && (ka  > qi - WIN_);
      bool a1 = (kb2 <= qi) && (kb2 > qi - WIN_);
      float s0 = a0 ? c0[r] * scale : -3.0e38f;
      float s1 = a1 ? c1[r] * scale : -3.0e38f;
      float pm = fmaxf(s0, s1);
      pm = fmaxf(pm, __shfl_xor(pm, 8, 16));
      pm = fmaxf(pm, __shfl_xor(pm, 4, 16));
      pm = fmaxf(pm, __shfl_xor(pm, 2, 16));
      pm = fmaxf(pm, __shfl_xor(pm, 1, 16));
      float mnew = fmaxf(mx[r], pm);
      float fac = __expf(mx[r] - mnew);
      float p0 = a0 ? __expf(s0 - mnew) : 0.0f;
      float p1 = a1 ? __expf(s1 - mnew) : 0.0f;
      float ps = p0 + p1;
      ps += __shfl_xor(ps, 8, 16);
      ps += __shfl_xor(ps, 4, 16);
      ps += __shfl_xor(ps, 2, 16);
      ps += __shfl_xor(ps, 1, 16);
      lsum[r] = lsum[r] * fac + ps;
      mx[r] = mnew;
      o0[r] *= fac; o1[r] *= fac; o2[r] *= fac; o3[r] *= fac;
      int mrow = r + hlf * 8;
      Psh[wib][mrow * 32 + l16]      = (_Float16)p0;
      Psh[wib][mrow * 32 + 16 + l16] = (_Float16)p1;
    }

    const _Float16* pp = &Psh[wib][l16 * 32 + hlf * 16];
    h16_t aP = load_a_frag(pp);
    {
      const _Float16* vp0 = Vbase + (size_t)(0 * 16 + l16) * S_ + k0 + hlf * 16;
      const _Float16* vp1 = Vbase + (size_t)(1 * 16 + l16) * S_ + k0 + hlf * 16;
      const _Float16* vp2 = Vbase + (size_t)(2 * 16 + l16) * S_ + k0 + hlf * 16;
      const _Float16* vp3 = Vbase + (size_t)(3 * 16 + l16) * S_ + k0 + hlf * 16;
      o0 = wmma_f16(aP, *(const h16_t*)vp0, o0);
      o1 = wmma_f16(aP, *(const h16_t*)vp1, o1);
      o2 = wmma_f16(aP, *(const h16_t*)vp2, o2);
      o3 = wmma_f16(aP, *(const h16_t*)vp3, o3);
    }
  }

#pragma unroll
  for (int r = 0; r < 8; ++r) {
    float inv = (lsum[r] > 0.0f) ? 1.0f / lsum[r] : 0.0f;
    int m = r + hlf * 8;
    size_t base = ((size_t)b * S_ + (q0 + m)) * D_ + h * HD_;
    CTX[base + 0 * 16 + l16] = (_Float16)(o0[r] * inv);
    CTX[base + 1 * 16 + l16] = (_Float16)(o1[r] * inv);
    CTX[base + 2 * 16 + l16] = (_Float16)(o2[r] * inv);
    CTX[base + 3 * 16 + l16] = (_Float16)(o3[r] * inv);
  }
}

// ---------------- GEMM 2: out = CTX @ Wp + bp  (fp32 output) ----------------

__global__ void __launch_bounds__(256) k_gemm_out(const _Float16* __restrict__ CTX,
                                                  const _Float16* __restrict__ WpT,
                                                  const float* __restrict__ bias,
                                                  float* __restrict__ out) {
  __shared__ _Float16 Bsh[2][64 * 32];
  const int NBT = D_ / 64;                            // 16
  int bm = blockIdx.x / NBT, bn = blockIdx.x % NBT;
  int n0 = bn * 64;
  int t = threadIdx.x;
  int lane = t & 31;
  int hlf = lane >> 4, l16 = lane & 15;
  int m0 = bm * 128 + (t >> 5) * 16;

  const _Float16* bsrc = WpT + (size_t)(n0 + (t >> 2)) * D_ + (t & 3) * 8;
  _Float16* bdst0 = &Bsh[0][t * 8];
  _Float16* bdst1 = &Bsh[1][t * 8];
  const _Float16* arow = CTX + (size_t)(m0 + l16) * D_ + hlf * 16;

  stage16(bsrc, bdst0);
  stage_wait();
  __syncthreads();

  f8_t acc[4] = {};
  for (int kb = 0; kb < D_; kb += 32) {
    int cur = (kb >> 5) & 1;
    if (kb + 32 < D_) stage16(bsrc + kb + 32, cur ? bdst0 : bdst1);
    const _Float16* Bcur = Bsh[cur];
    h16_t a = load_a_frag(arow + kb);
#pragma unroll
    for (int j = 0; j < 4; ++j) {
      h16_t b = *(const h16_t*)&Bcur[(j * 16 + l16) * 32 + hlf * 16];
      acc[j] = wmma_f16(a, b, acc[j]);
    }
    stage_wait();
    __syncthreads();
  }

#pragma unroll
  for (int j = 0; j < 4; ++j) {
    int n = n0 + j * 16 + l16;
    float bn2 = bias[n];
#pragma unroll
    for (int r = 0; r < 8; ++r) {
      int m = m0 + r + hlf * 8;
      out[(size_t)m * D_ + n] = acc[j][r] + bn2;
    }
  }
}

// ---------------- host launch ----------------

extern "C" void kernel_launch(void* const* d_in, const int* in_sizes, int n_in,
                              void* d_out, int out_size, void* d_ws, size_t ws_size,
                              hipStream_t stream) {
  const float* hidden = (const float*)d_in[0];
  // d_in[1] = position_ids (arange, unused: position == s)
  const float* Wa = (const float*)d_in[2];
  const float* ba = (const float*)d_in[3];
  const float* Wp = (const float*)d_in[4];
  const float* bp = (const float*)d_in[5];
  float* out = (float*)d_out;

  char* ws = (char*)d_ws;
  _Float16* X16 = (_Float16*)(ws + 0);          //  8 MB  [4096,1024]
  _Float16* WaT = (_Float16*)(ws + 8388608);    //  6 MB  [3072,1024]
  _Float16* WpT = (_Float16*)(ws + 14680064);   //  2 MB  [1024,1024]
  _Float16* QKV = (_Float16*)(ws + 16777216);   // 24 MB  [3,B,H,S,HD]
  _Float16* VT  = (_Float16*)(ws + 41943040);   //  8 MB  [B,H,HD,S]
  _Float16* CTX = (_Float16*)(ws + 50331648);   //  8 MB  [4096,1024]

  k_cvt<<<(M_ * D_) / 256, 256, 0, stream>>>(hidden, X16, M_ * D_);
  k_transpose_cvt<<<(D_ * 3 * D_) / 256, 256, 0, stream>>>(Wa, WaT, D_, 3 * D_);
  k_transpose_cvt<<<(D_ * D_) / 256, 256, 0, stream>>>(Wp, WpT, D_, D_);
  k_gemm_qkv<<<(M_ / 128) * ((3 * D_) / 64), 256, 0, stream>>>(X16, WaT, ba, QKV);
  k_rope<<<(2 * B_ * H_ * S_ * 32) / 256, 256, 0, stream>>>(QKV);
  k_vt<<<(B_ * H_ * S_ * HD_) / 256, 256, 0, stream>>>(QKV, VT);
  k_attn<<<(B_ * H_ * (S_ / 16)) / 4, 128, 0, stream>>>(QKV, VT, CTX);
  k_gemm_out<<<(M_ / 128) * (D_ / 64), 256, 0, stream>>>(CTX, WpT, bp, out);
}
